// BayeshERG_11802570129439
// MI455X (gfx1250) — matrix operand order repaired and checked by reading, not compiled, not run on hardware
//
#include <hip/hip_runtime.h>
#include <stdint.h>

// ---------------------------------------------------------------------------
// BayeshERG DMPNN forward for MI455X (gfx1250), wave32 + WMMA bf16.
//   N=450000 nodes, E=1000000 edges (reverse pairs e^1), T=7 steps
//   hidden = 128, node feat = 74, edge feat = 12
// Memory-bound (~20 GB moved). GEMMs: v_wmma_f32_16x16x32_bf16, fp32 accum.
// Weights pre-converted to transposed/padded bf16 once, then staged to LDS
// with GLOBAL_LOAD_ASYNC_TO_LDS_B128 overlapped with activation staging.
// ---------------------------------------------------------------------------

#define N_NODES 450000
#define N_EDGES 1000000
#define FNID    74
#define FEID    12
#define EH      128
#define NSTEPS  7

#define KP_INIT 96      // 86 -> padded to 3 K-tiles
#define KP_STEP 128     // 4 K-tiles
#define KP_LAST 224     // 202 -> padded to 7 K-tiles

typedef __attribute__((ext_vector_type(16))) __bf16 v16bf;
typedef __attribute__((ext_vector_type(8)))  __bf16 v8bf;
typedef __attribute__((ext_vector_type(8)))  float  v8f;

// ---------------- deterministic per-element uniform hash -------------------
__device__ __forceinline__ float hash_u(uint32_t e, uint32_t k, uint32_t strm) {
    uint32_t h = e * 0x9E3779B1u + k * 0x85EBCA77u + strm * 0xC2B2AE3Du;
    h ^= h >> 16; h *= 0x7FEB352Du;
    h ^= h >> 15; h *= 0x846CA68Bu;
    h ^= h >> 16;
    return ((h >> 8) + 0.5f) * (1.0f / 16777216.0f);   // (0,1)
}

// concrete dropout keep-factor: (1 - sigmoid((c0 + logit(u)) / temp)) / (1 - p)
__device__ __forceinline__ float ckeep(float u, float c0, float inv1mp) {
    float l = (c0 + __logf(u + 1e-7f) - __logf(1.0f - u + 1e-7f)) * 10.0f;
    float s = 1.0f / (1.0f + __expf(-l));
    return (1.0f - s) * inv1mp;
}

// ---------------- WMMA fragment builders (LDS, bf16) -----------------------
// A 16x32: lane(m=L&15, kh=L>>4): elems 0..7 = K[k0..k0+7], 8..15 = K[k0+16..]
__device__ __forceinline__ v16bf load_a_frag(const __bf16* p) {
    v8bf lo = *(const v8bf*)(p);
    v8bf hi = *(const v8bf*)(p + 16);
    v16bf r;
#pragma unroll
    for (int i = 0; i < 8; ++i) { r[i] = lo[i]; r[i + 8] = hi[i]; }
    return r;
}
// B 32x16: lane(n=L&15, kh=L>>4): elems 0..7 = K[k0..k0+7], 8..15 = K[k0+16..]
__device__ __forceinline__ v16bf load_b_frag(const __bf16* p) {
    v8bf lo = *(const v8bf*)(p);
    v8bf hi = *(const v8bf*)(p + 16);
    v16bf r;
#pragma unroll
    for (int i = 0; i < 8; ++i) { r[i] = lo[i]; r[i + 8] = hi[i]; }
    return r;
}

// ---------------- async bf16 weight image -> LDS ---------------------------
// Layouts are identical flat byte streams, so this is a pure async copy.
// GVS addressing: mem = SADDR + VADDR; LDS addr = LDS_BASE + VDST. 16B/lane.
__device__ __forceinline__ void async_wload(const __bf16* __restrict__ Wt,
                                            uint32_t nbytes, __bf16* Ws, int t) {
    uint32_t base = (uint32_t)(uintptr_t)Ws;           // LDS byte offset
    for (uint32_t off = (uint32_t)t * 16u; off < nbytes; off += 256u * 16u) {
        uint32_t lds = base + off;
        asm volatile("global_load_async_to_lds_b128 %0, %1, %2"
                     :: "v"(lds), "v"(off), "s"((const void*)Wt)
                     : "memory");
    }
}
__device__ __forceinline__ void async_wait_all() {
    asm volatile("s_wait_asynccnt 0" ::: "memory");
}

// ---------------- one-shot: transpose + pad + bf16-convert weights ---------
__global__ void wconv_kernel(const float* __restrict__ Wi,  // [86][128]
                             const float* __restrict__ Wu,  // [128][128]
                             const float* __restrict__ Wl,  // [202][128]
                             __bf16* __restrict__ WtI,      // [128][96]
                             __bf16* __restrict__ WtU,      // [128][128]
                             __bf16* __restrict__ WtL) {    // [128][224]
    int i = blockIdx.x * 256 + threadIdx.x;
    const int SI = 128 * KP_INIT, SU = 128 * KP_STEP, SL = 128 * KP_LAST;
    if (i < SI) {
        int n = i / KP_INIT, k = i - n * KP_INIT;
        WtI[i] = (__bf16)((k < FNID + FEID) ? Wi[k * EH + n] : 0.0f);
    } else if (i < SI + SU) {
        int j = i - SI, n = j >> 7, k = j & 127;
        WtU[j] = (__bf16)Wu[k * EH + n];
    } else if (i < SI + SU + SL) {
        int j = i - SI - SU, n = j / KP_LAST, k = j - n * KP_LAST;
        WtL[j] = (__bf16)((k < FNID + EH) ? Wl[k * EH + n] : 0.0f);
    }
}

// ---------------- zero node accumulator ------------------------------------
__global__ void zero_kernel(float4* __restrict__ p, long n4) {
    long i = (long)blockIdx.x * blockDim.x + threadIdx.x;
    if (i < n4) p[i] = make_float4(0.f, 0.f, 0.f, 0.f);
}

// ---------------- segment_sum: scatter edge states to dst nodes ------------
__global__ void scatter_kernel(const float* __restrict__ et,
                               const int* __restrict__ dst,
                               float* __restrict__ nacc) {
    long i = (long)blockIdx.x * blockDim.x + threadIdx.x;   // over E*EH
    if (i < (long)N_EDGES * EH) {
        int e = (int)(i >> 7), k = (int)(i & 127);
        float v = et[i];
        __hip_atomic_fetch_add(&nacc[(size_t)dst[e] * EH + k], v,
                               __ATOMIC_RELAXED, __HIP_MEMORY_SCOPE_AGENT);
    }
}

// ---------------- e0 = relu(CD(concat(nf[src], ef)) @ W_init) --------------
__global__ void init_kernel(const float* __restrict__ n_feat,
                            const float* __restrict__ e_feat,
                            const __bf16* __restrict__ Wt,   // [128][96] bf16
                            const float* __restrict__ p_logit,
                            const int* __restrict__ src,
                            const int* __restrict__ seedp,
                            float* __restrict__ e0) {
    constexpr int KP = KP_INIT;
    extern __shared__ __align__(16) __bf16 smem[];
    __bf16* Ws = smem;                 // [128 n][KP k]
    __bf16* Xs = smem + 128 * KP;      // [128 edge][KP k]
    const int t = threadIdx.x;
    const int ebase = blockIdx.x * 128;
    const bool full = (ebase + 128 <= N_EDGES);

    async_wload(Wt, 128 * KP * sizeof(__bf16), Ws, t);   // overlap w/ staging

    const uint32_t seed = (uint32_t)seedp[0];
    const uint32_t strm = seed * 0x9E3779B9u;             // keys[0]
    const float pl = p_logit[0];
    const float p  = 1.0f / (1.0f + __expf(-pl));
    const float c0 = __logf(p + 1e-7f) - __logf(1.0f - p + 1e-7f);
    const float i1 = 1.0f / (1.0f - p);

    for (int i = t; i < 128 * KP; i += 256) {             // X rows, k-coalesced
        int el = i / KP, k = i - el * KP;
        int e = ebase + el;
        float x = 0.0f;
        if (e < N_EDGES && k < FNID + FEID) {
            x = (k < FNID) ? n_feat[(size_t)src[e] * FNID + k]
                           : e_feat[(size_t)e * FEID + (k - FNID)];
            x *= ckeep(hash_u((uint32_t)e, (uint32_t)k, strm), c0, i1);
        }
        Xs[el * KP + k] = (__bf16)x;
    }
    async_wait_all();
    __syncthreads();

    const int wave = t >> 5, lane = t & 31;
    const int m = lane & 15, kh = lane >> 4;
    const __bf16* xrow = &Xs[(wave * 16 + m) * KP];
    v16bf a[3];
#pragma unroll
    for (int kt = 0; kt < 3; ++kt) a[kt] = load_a_frag(xrow + kt * 32 + kh * 8);

#pragma unroll
    for (int nt = 0; nt < 8; ++nt) {
        int ncol = nt * 16 + m;
        const __bf16* wrow = &Ws[ncol * KP];
        v8f c = {};
#pragma unroll
        for (int kt = 0; kt < 3; ++kt) {
            v16bf b = load_b_frag(wrow + kt * 32 + kh * 8);
            c = __builtin_amdgcn_wmma_f32_16x16x32_bf16(false, a[kt], false, b,
                                                        (short)0, c, false, false);
        }
        float* drow = &e0[(size_t)(ebase + wave * 16 + 8 * kh) * EH + ncol];
        if (full) {
#pragma unroll
            for (int v = 0; v < 8; ++v) drow[(size_t)v * EH] = fmaxf(c[v], 0.0f);
        } else {
#pragma unroll
            for (int v = 0; v < 8; ++v) {
                int e = ebase + wave * 16 + v + 8 * kh;
                if (e < N_EDGES) e0[(size_t)e * EH + ncol] = fmaxf(c[v], 0.0f);
            }
        }
    }
}

// ---------------- e_new = relu(e0 + CD(nacc[src] - e_t[e^1]) @ W_eupd) -----
__global__ void step_kernel(const float* __restrict__ nacc,
                            const float* __restrict__ ecur,
                            const float* __restrict__ e0,
                            const __bf16* __restrict__ Wt,   // [128][128] bf16
                            const float* __restrict__ p_logit,
                            const int* __restrict__ src,
                            const int* __restrict__ seedp,
                            float* __restrict__ enew,
                            int stepid) {
    constexpr int KP = KP_STEP;
    extern __shared__ __align__(16) __bf16 smem[];
    __bf16* Ws = smem;                 // [128 n][128 k]
    __bf16* Xs = smem + 128 * KP;      // [128 edge][128 k]
    const int t = threadIdx.x;
    const int ebase = blockIdx.x * 128;
    const bool full = (ebase + 128 <= N_EDGES);

    async_wload(Wt, 128 * KP * sizeof(__bf16), Ws, t);

    const uint32_t seed = (uint32_t)seedp[0];
    const uint32_t strm = seed * 0x9E3779B9u + (uint32_t)stepid * 0x85EBCA6Bu;
    const float pl = p_logit[0];
    const float p  = 1.0f / (1.0f + __expf(-pl));
    const float c0 = __logf(p + 1e-7f) - __logf(1.0f - p + 1e-7f);
    const float i1 = 1.0f / (1.0f - p);

    for (int i = t; i < 128 * KP; i += 256) {
        int el = i >> 7, k = i & 127;
        int e = ebase + el;
        float x = 0.0f;
        if (e < N_EDGES) {
            float xr = nacc[(size_t)src[e] * EH + k] - ecur[(size_t)(e ^ 1) * EH + k];
            x = xr * ckeep(hash_u((uint32_t)e, (uint32_t)k, strm), c0, i1);
        }
        Xs[el * KP + k] = (__bf16)x;
    }
    async_wait_all();
    __syncthreads();

    const int wave = t >> 5, lane = t & 31;
    const int m = lane & 15, kh = lane >> 4;
    const __bf16* xrow = &Xs[(wave * 16 + m) * KP];
    v16bf a[4];
#pragma unroll
    for (int kt = 0; kt < 4; ++kt) a[kt] = load_a_frag(xrow + kt * 32 + kh * 8);

#pragma unroll
    for (int nt = 0; nt < 8; ++nt) {
        int ncol = nt * 16 + m;
        const size_t rowbase = (size_t)(ebase + wave * 16 + 8 * kh) * EH + ncol;
        v8f c;
        if (full) {
#pragma unroll
            for (int v = 0; v < 8; ++v) c[v] = e0[rowbase + (size_t)v * EH];
        } else {
#pragma unroll
            for (int v = 0; v < 8; ++v) {
                int e = ebase + wave * 16 + v + 8 * kh;
                c[v] = (e < N_EDGES) ? e0[(size_t)e * EH + ncol] : 0.0f;
            }
        }
        const __bf16* wrow = &Ws[ncol * KP];
#pragma unroll
        for (int kt = 0; kt < 4; ++kt) {
            v16bf b = load_b_frag(wrow + kt * 32 + kh * 8);
            c = __builtin_amdgcn_wmma_f32_16x16x32_bf16(false, a[kt], false, b,
                                                        (short)0, c, false, false);
        }
        if (full) {
#pragma unroll
            for (int v = 0; v < 8; ++v) enew[rowbase + (size_t)v * EH] = fmaxf(c[v], 0.0f);
        } else {
#pragma unroll
            for (int v = 0; v < 8; ++v) {
                int e = ebase + wave * 16 + v + 8 * kh;
                if (e < N_EDGES) enew[(size_t)e * EH + ncol] = fmaxf(c[v], 0.0f);
            }
        }
    }
}

// ---------------- out = relu(CD(concat(nf, ff)) @ W_last) ------------------
__global__ void final_kernel(const float* __restrict__ n_feat,
                             const float* __restrict__ ff,
                             const __bf16* __restrict__ Wt,  // [128][224] bf16
                             const float* __restrict__ p_logit,
                             const int* __restrict__ seedp,
                             float* __restrict__ out) {
    constexpr int KP = KP_LAST, KR = FNID + EH;              // 224, 202
    extern __shared__ __align__(16) __bf16 smem[];
    __bf16* Ws = smem;                 // [128 n][KP k]
    __bf16* Xs = smem + 128 * KP;      // [128 node][KP k]
    const int t = threadIdx.x;
    const int nbase = blockIdx.x * 128;
    const bool full = (nbase + 128 <= N_NODES);

    async_wload(Wt, 128 * KP * sizeof(__bf16), Ws, t);

    const uint32_t seed = (uint32_t)seedp[0];
    const uint32_t strm = seed * 0x9E3779B9u + (uint32_t)(NSTEPS + 1) * 0x85EBCA6Bu;
    const float pl = p_logit[0];
    const float p  = 1.0f / (1.0f + __expf(-pl));
    const float c0 = __logf(p + 1e-7f) - __logf(1.0f - p + 1e-7f);
    const float i1 = 1.0f / (1.0f - p);

    for (int i = t; i < 128 * KP; i += 256) {
        int nl = i / KP, k = i - nl * KP;
        int node = nbase + nl;
        float x = 0.0f;
        if (node < N_NODES && k < KR) {
            x = (k < FNID) ? n_feat[(size_t)node * FNID + k]
                           : ff[(size_t)node * EH + (k - FNID)];
            x *= ckeep(hash_u((uint32_t)node, (uint32_t)k, strm), c0, i1);
        }
        Xs[nl * KP + k] = (__bf16)x;
    }
    async_wait_all();
    __syncthreads();

    const int wave = t >> 5, lane = t & 31;
    const int m = lane & 15, kh = lane >> 4;
    const __bf16* xrow = &Xs[(wave * 16 + m) * KP];
    v16bf a[7];
#pragma unroll
    for (int kt = 0; kt < 7; ++kt) a[kt] = load_a_frag(xrow + kt * 32 + kh * 8);

#pragma unroll
    for (int nt = 0; nt < 8; ++nt) {
        int ncol = nt * 16 + m;
        const __bf16* wrow = &Ws[ncol * KP];
        v8f c = {};
#pragma unroll
        for (int kt = 0; kt < 7; ++kt) {
            v16bf b = load_b_frag(wrow + kt * 32 + kh * 8);
            c = __builtin_amdgcn_wmma_f32_16x16x32_bf16(false, a[kt], false, b,
                                                        (short)0, c, false, false);
        }
        if (full) {
            float* drow = &out[(size_t)(nbase + wave * 16 + 8 * kh) * EH + ncol];
#pragma unroll
            for (int v = 0; v < 8; ++v) drow[(size_t)v * EH] = fmaxf(c[v], 0.0f);
        } else {
#pragma unroll
            for (int v = 0; v < 8; ++v) {
                int node = nbase + wave * 16 + v + 8 * kh;
                if (node < N_NODES) out[(size_t)node * EH + ncol] = fmaxf(c[v], 0.0f);
            }
        }
    }
}

// ---------------------------------------------------------------------------
extern "C" void kernel_launch(void* const* d_in, const int* in_sizes, int n_in,
                              void* d_out, int out_size, void* d_ws, size_t ws_size,
                              hipStream_t stream) {
    const float* n_feat = (const float*)d_in[0];
    const float* e_feat = (const float*)d_in[1];
    const float* W_init = (const float*)d_in[2];
    const float* W_eupd = (const float*)d_in[3];
    const float* W_last = (const float*)d_in[4];
    const float* p_init = (const float*)d_in[5];
    const float* p_eupd = (const float*)d_in[6];
    const float* p_last = (const float*)d_in[7];
    const int*   src    = (const int*)d_in[8];
    const int*   dst    = (const int*)d_in[9];
    const int*   seedp  = (const int*)d_in[10];
    float* out = (float*)d_out;

    // workspace: e0 | ping | pong | node accumulator | bf16 weight images
    float* e0   = (float*)d_ws;
    float* eA   = e0 + (size_t)N_EDGES * EH;
    float* eB   = eA + (size_t)N_EDGES * EH;
    float* nacc = eB + (size_t)N_EDGES * EH;
    __bf16* WtI = (__bf16*)(nacc + (size_t)N_NODES * EH);   // 16B-aligned
    __bf16* WtU = WtI + 128 * KP_INIT;
    __bf16* WtL = WtU + 128 * KP_STEP;

    const dim3 B(256);
    const int edgeBlocks = (N_EDGES + 127) / 128;             // 7813
    const int nodeBlocks = (N_NODES + 127) / 128;             // 3516
    const int scatBlocks = (int)(((long)N_EDGES * EH + 255) / 256);
    const long n4        = (long)N_NODES * EH / 4;
    const int zeroBlocks = (int)((n4 + 255) / 256);
    const int wconvTotal = 128 * (KP_INIT + KP_STEP + KP_LAST);
    const int wconvBlocks = (wconvTotal + 255) / 256;

    const size_t lds_init  = (size_t)2 * 128 * KP_INIT * sizeof(__bf16); //  48 KB
    const size_t lds_step  = (size_t)2 * 128 * KP_STEP * sizeof(__bf16); //  64 KB
    const size_t lds_final = (size_t)2 * 128 * KP_LAST * sizeof(__bf16); // 112 KB

    wconv_kernel<<<wconvBlocks, B, 0, stream>>>(W_init, W_eupd, W_last,
                                                WtI, WtU, WtL);

    init_kernel<<<edgeBlocks, B, lds_init, stream>>>(n_feat, e_feat, WtI,
                                                     p_init, src, seedp, e0);

    const float* cur = e0;
    float* pingpong[2] = { eA, eB };
    for (int tstep = 0; tstep < NSTEPS; ++tstep) {
        zero_kernel<<<zeroBlocks, B, 0, stream>>>((float4*)nacc, n4);
        scatter_kernel<<<scatBlocks, B, 0, stream>>>(cur, dst, nacc);
        float* nxt = pingpong[tstep & 1];
        step_kernel<<<edgeBlocks, B, lds_step, stream>>>(nacc, cur, e0, WtU,
                                                         p_eupd, src, seedp, nxt,
                                                         tstep + 1);
        cur = nxt;
    }

    zero_kernel<<<zeroBlocks, B, 0, stream>>>((float4*)nacc, n4);
    scatter_kernel<<<scatBlocks, B, 0, stream>>>(cur, dst, nacc);
    final_kernel<<<nodeBlocks, B, lds_final, stream>>>(n_feat, nacc, WtL,
                                                       p_last, seedp, out);
}
